// Luong_Encoder_42528766165274
// MI455X (gfx1250) — matrix operand order repaired and checked by reading, not compiled, not run on hardware
//
#include <hip/hip_runtime.h>
#include <hip/hip_bf16.h>

#define EMB 512
#define HID 1024
#define H2  512
#define SEQLEN 512
#define BATCH 256
#define XH_ROWS 32

typedef __attribute__((ext_vector_type(16))) __bf16 v16bf;
typedef __attribute__((ext_vector_type(8)))  __bf16 v8bf;
typedef __attribute__((ext_vector_type(8)))  float  v8f;

union ABFrag { v16bf v; v8bf h[2]; };

// A-matrix fragment (16x32 bf16, M x K): lanes 0-15 -> rows M=0..15 with K = {k0..k0+7, k0+16..k0+23},
// lanes 16-31 -> same rows, K shifted by 8. base points at row-major M x ldk tile.
__device__ __forceinline__ v16bf load_a(const __bf16* base, int ldk, int lane, int kt) {
    int m  = lane & 15;
    int k0 = kt * 32 + ((lane >> 4) << 3);
    ABFrag f;
    f.h[0] = *reinterpret_cast<const v8bf*>(base + (size_t)m * ldk + k0);
    f.h[1] = *reinterpret_cast<const v8bf*>(base + (size_t)m * ldk + k0 + 16);
    return f.v;
}

// B-matrix fragment (32x16 bf16, K x N): lane n<16 holds column n with K = {k0..k0+7, k0+16..k0+23}; lanes 16-31 shifted by 8.
// wT is the weight stored transposed: [N][K] row-major bf16, so each lane's 16 K-values are two contiguous runs of 8.
__device__ __forceinline__ v16bf load_b(const __bf16* wT, int ldk, int lane, int nt, int kt) {
    int n  = nt * 16 + (lane & 15);
    int k0 = kt * 32 + ((lane >> 4) << 3);
    ABFrag f;
    f.h[0] = *reinterpret_cast<const v8bf*>(wT + (size_t)n * ldk + k0);
    f.h[1] = *reinterpret_cast<const v8bf*>(wT + (size_t)n * ldk + k0 + 16);
    return f.v;
}

__device__ __forceinline__ v8f wmma_bf16(v16bf a, v16bf b, v8f c) {
    // (neg_a, A, neg_b, B, c_mod, C, reuse_a, reuse_b)
    return __builtin_amdgcn_wmma_f32_16x16x32_bf16(false, a, false, b, (short)0, c, false, false);
}

// ---- weight transpose + f32->bf16: wT[n*K + k] = (bf16) w[k*N + n] ----
__global__ void convert_wT(const float* __restrict__ w, __bf16* __restrict__ wT, int K, int N) {
    int tid = blockIdx.x * 256 + threadIdx.x;
    if (tid >= K * N) return;
    int n = tid / K;
    int k = tid - n * K;
    wT[tid] = (__bf16)w[(size_t)k * N + n];
}

// ---- h0 = 0 (both f32 and bf16 copies) ----
__global__ void zero_h(float* __restrict__ hf, __bf16* __restrict__ hb, int n) {
    int tid = blockIdx.x * 256 + threadIdx.x;
    if (tid >= n) return;
    hf[tid] = 0.0f;
    hb[tid] = (__bf16)0.0f;
}

// ---- Phase 1: xh = ReLU(seq @ w1x + b1x) @ w2x + b2x, written into d_out ----
// grid = (B*S)/32 blocks, 256 threads (8 waves). Block owns 32 (b,s) rows:
// each B-fragment load is reused by two M-tiles -> halves L2 weight traffic.
__global__ __launch_bounds__(256) void xh_kernel(
    const float* __restrict__ seq,
    const __bf16* __restrict__ w1xT, const float* __restrict__ b1x,
    const __bf16* __restrict__ w2xT, const float* __restrict__ b2x,
    float* __restrict__ out) {
    __shared__ __bf16 Abf[XH_ROWS * EMB];   // 32 KB: seq tile in bf16
    __shared__ __bf16 hid[XH_ROWS * H2];    // 32 KB: ReLU intermediate in bf16

    const int tid  = threadIdx.x;
    const int lane = tid & 31;
    const int wave = tid >> 5;
    const long long row0 = (long long)blockIdx.x * XH_ROWS;   // flat (b*S + s) row base

    // cooperative load + f32->bf16 convert of the 32x512 A tile (coalesced)
    #pragma unroll
    for (int i = 0; i < 64; ++i) {
        int idx = tid + i * 256;          // 0..16383
        int m   = idx >> 9;
        int k   = idx & 511;
        Abf[idx] = (__bf16)seq[(row0 + m) * EMB + k];
    }
    __syncthreads();

    // GEMM1: hid = ReLU(A @ w1x + b1x); 2 M-tiles x 4 N-tiles per wave; K=512 -> 16 k-steps
    {
        v8f acc[4][2] = {};
        for (int kt = 0; kt < EMB / 32; ++kt) {
            v16bf a0 = load_a(Abf,            EMB, lane, kt);
            v16bf a1 = load_a(Abf + 16 * EMB, EMB, lane, kt);
            #pragma unroll
            for (int j = 0; j < 4; ++j) {
                v16bf b = load_b(w1xT, EMB, lane, wave * 4 + j, kt);
                acc[j][0] = wmma_bf16(a0, b, acc[j][0]);
                acc[j][1] = wmma_bf16(a1, b, acc[j][1]);
            }
        }
        #pragma unroll
        for (int j = 0; j < 4; ++j) {
            int gn = (wave * 4 + j) * 16 + (lane & 15);
            float bias = b1x[gn];
            int mbase = (lane >> 4) * 8;
            #pragma unroll
            for (int mt = 0; mt < 2; ++mt) {
                #pragma unroll
                for (int i = 0; i < 8; ++i) {
                    float v = acc[j][mt][i] + bias;
                    hid[(mt * 16 + mbase + i) * H2 + gn] = (__bf16)(v > 0.0f ? v : 0.0f);
                }
            }
        }
    }
    __syncthreads();

    // GEMM2: out = hid @ w2x + b2x; N=1024 in two passes of (2 M-tiles x 4 N-tiles)/wave; K=512
    #pragma unroll 1
    for (int nc = 0; nc < 2; ++nc) {
        v8f acc[4][2] = {};
        for (int kt = 0; kt < H2 / 32; ++kt) {
            v16bf a0 = load_a(hid,           H2, lane, kt);
            v16bf a1 = load_a(hid + 16 * H2, H2, lane, kt);
            #pragma unroll
            for (int j = 0; j < 4; ++j) {
                v16bf b = load_b(w2xT, H2, lane, wave * 8 + nc * 4 + j, kt);
                acc[j][0] = wmma_bf16(a0, b, acc[j][0]);
                acc[j][1] = wmma_bf16(a1, b, acc[j][1]);
            }
        }
        #pragma unroll
        for (int j = 0; j < 4; ++j) {
            int gn = (wave * 8 + nc * 4 + j) * 16 + (lane & 15);
            float bias = b2x[gn];
            int mbase = (lane >> 4) * 8;
            #pragma unroll
            for (int mt = 0; mt < 2; ++mt) {
                #pragma unroll
                for (int i = 0; i < 8; ++i) {
                    out[(row0 + mt * 16 + mbase + i) * HID + gn] = acc[j][mt][i] + bias;
                }
            }
        }
    }
}

// ---- Phase 2 (one launch per timestep): h = select(mask, tanh(xh_t + MLP(h)), h) ----
// grid = 16 blocks (one per 16 batch rows), 512 threads (16 waves) to shorten the
// per-step dependent WMMA chain (2 n-tiles/wave GEMM1, 4 n-tiles/wave GEMM2).
__global__ __launch_bounds__(512) void step_kernel(
    __bf16* __restrict__ hbf, float* __restrict__ hf,
    const __bf16* __restrict__ w1hT, const float* __restrict__ b1h,
    const __bf16* __restrict__ w2hT, const float* __restrict__ b2h,
    const int* __restrict__ lens, float* __restrict__ out, int t) {
    __shared__ __bf16 mid[16 * H2];    // 16 KB: ReLU intermediate in bf16

    const int tid  = threadIdx.x;
    const int lane = tid & 31;
    const int wave = tid >> 5;         // 0..15
    const int row0 = blockIdx.x * 16;  // batch row base

    // GEMM1: mid = ReLU(h @ w1h + b1h); K=1024 -> 32 k-steps; 2 n-tiles/wave
    {
        const __bf16* atile = hbf + (size_t)row0 * HID;
        v8f acc[2] = {};
        for (int kt = 0; kt < HID / 32; ++kt) {
            v16bf a = load_a(atile, HID, lane, kt);
            #pragma unroll
            for (int j = 0; j < 2; ++j) {
                v16bf b = load_b(w1hT, HID, lane, wave * 2 + j, kt);
                acc[j] = wmma_bf16(a, b, acc[j]);
            }
        }
        #pragma unroll
        for (int j = 0; j < 2; ++j) {
            int gn = (wave * 2 + j) * 16 + (lane & 15);
            float bias = b1h[gn];
            int mbase = (lane >> 4) * 8;
            #pragma unroll
            for (int i = 0; i < 8; ++i) {
                float v = acc[j][i] + bias;
                mid[(mbase + i) * H2 + gn] = (__bf16)(v > 0.0f ? v : 0.0f);
            }
        }
    }
    __syncthreads();

    // GEMM2 + epilogue: hh = mid @ w2h + b2h; h = mask ? tanh(xh_t + hh) : h
    {
        v8f acc[4] = {};
        for (int kt = 0; kt < H2 / 32; ++kt) {
            v16bf a = load_a(mid, H2, lane, kt);
            #pragma unroll
            for (int j = 0; j < 4; ++j) {
                v16bf b = load_b(w2hT, H2, lane, wave * 4 + j, kt);
                acc[j] = wmma_bf16(a, b, acc[j]);
            }
        }
        #pragma unroll
        for (int j = 0; j < 4; ++j) {
            int gn = (wave * 4 + j) * 16 + (lane & 15);
            float bias = b2h[gn];
            int mbase = (lane >> 4) * 8;
            #pragma unroll
            for (int i = 0; i < 8; ++i) {
                int brow = row0 + mbase + i;
                size_t hidx = (size_t)brow * HID + gn;
                size_t oidx = ((size_t)brow * SEQLEN + t) * HID + gn;
                float hh = acc[j][i] + bias;
                float nh = tanhf(out[oidx] + hh);       // out currently holds xh_t
                float hv = (t < lens[brow]) ? nh : hf[hidx];
                hf[hidx]  = hv;
                hbf[hidx] = (__bf16)hv;
                out[oidx] = hv;                          // overwrite xh_t with h_t
            }
        }
    }
}

extern "C" void kernel_launch(void* const* d_in, const int* in_sizes, int n_in,
                              void* d_out, int out_size, void* d_ws, size_t ws_size,
                              hipStream_t stream) {
    const float* seq  = (const float*)d_in[0];
    const int*   lens = (const int*)  d_in[1];
    const float* w1x  = (const float*)d_in[2];
    const float* b1x  = (const float*)d_in[3];
    const float* w2x  = (const float*)d_in[4];
    const float* b2x  = (const float*)d_in[5];
    const float* w1h  = (const float*)d_in[6];
    const float* b1h  = (const float*)d_in[7];
    const float* w2h  = (const float*)d_in[8];
    const float* b2h  = (const float*)d_in[9];
    float* out = (float*)d_out;

    // workspace layout (all offsets well-aligned): ~5 MB total
    char* ws = (char*)d_ws;
    __bf16* w1xT = (__bf16*)ws; ws += (size_t)H2  * EMB * 2;   // [512][512]
    __bf16* w2xT = (__bf16*)ws; ws += (size_t)HID * H2  * 2;   // [1024][512]
    __bf16* w1hT = (__bf16*)ws; ws += (size_t)H2  * HID * 2;   // [512][1024]
    __bf16* w2hT = (__bf16*)ws; ws += (size_t)HID * H2  * 2;   // [1024][512]
    __bf16* hbf  = (__bf16*)ws; ws += (size_t)BATCH * HID * 2; // h state, bf16
    float*  hf   = (float*) ws; ws += (size_t)BATCH * HID * 4; // h state, f32

    convert_wT<<<(EMB * H2  + 255) / 256, 256, 0, stream>>>(w1x, w1xT, EMB, H2);
    convert_wT<<<(H2  * HID + 255) / 256, 256, 0, stream>>>(w2x, w2xT, H2,  HID);
    convert_wT<<<(HID * H2  + 255) / 256, 256, 0, stream>>>(w1h, w1hT, HID, H2);
    convert_wT<<<(H2  * HID + 255) / 256, 256, 0, stream>>>(w2h, w2hT, H2,  HID);
    zero_h<<<(BATCH * HID + 255) / 256, 256, 0, stream>>>(hf, hbf, BATCH * HID);

    // Phase 1: xh for all (b,s) rows -> d_out
    xh_kernel<<<(BATCH * SEQLEN) / XH_ROWS, 256, 0, stream>>>(seq, w1xT, b1x, w2xT, b2x, out);

    // Phase 2: sequential scan, one kernel per timestep (stream-ordered)
    for (int t = 0; t < SEQLEN; ++t) {
        step_kernel<<<16, 512, 0, stream>>>(hbf, hf, w1hT, b1h, w2hT, b2h, lens, out, t);
    }
}
